// Nucleus_67190468378720
// MI455X (gfx1250) — matrix operand order repaired and checked by reading, not compiled, not run on hardware
//
#include <hip/hip_runtime.h>

constexpr int Vv   = 50257;
constexpr int VvP  = 50304;            // padded vocab rows (393 * 128)
constexpr int Dd   = 1024;
constexpr int Hh   = 16;
constexpr int FFd  = 4096;
constexpr int Nn   = 2000;
constexpr int Kk   = 20;
constexpr int Bb   = 4;
constexpr int Ss   = 256;
constexpr int BSr  = Bb * Ss;          // 1024 activation rows

#define NEG_INF (-__builtin_inff())

typedef __attribute__((ext_vector_type(16))) __bf16 v16bf;
typedef __attribute__((ext_vector_type(8)))  float  v8f;

union Frag { uint4 q[2]; v16bf v; };   // 16 bf16 = 32B, loaded as 2x b128

static __device__ __forceinline__ unsigned short f2bf(float x) {
  union { float f; unsigned int u; } c; c.f = x;
  unsigned int u = c.u;
  unsigned int r = u + 0x7FFFu + ((u >> 16) & 1u);   // round-to-nearest-even
  return (unsigned short)(r >> 16);
}

// ---------------- bulk f32 -> bf16 conversion (4 elems / thread) ------------
// n4tot may exceed n4src: the tail is zero-filled (used to pad dec_w rows).
__global__ __launch_bounds__(256) void k_cvt(const float* __restrict__ src,
                                             unsigned short* __restrict__ dst,
                                             int n4src, int n4tot) {
  const int i = blockIdx.x * 256 + threadIdx.x;
  if (i >= n4tot) return;
  ushort4 o;
  if (i < n4src) {
    const float4 f = reinterpret_cast<const float4*>(src)[i];
    o.x = f2bf(f.x); o.y = f2bf(f.y); o.z = f2bf(f.z); o.w = f2bf(f.w);
  } else {
    o.x = 0; o.y = 0; o.z = 0; o.w = 0;
  }
  reinterpret_cast<ushort4*>(dst)[i] = o;
}

// ---------------- embedding gather ----------------
__global__ __launch_bounds__(256) void k_embed(const int* __restrict__ inp,
                                               const float* __restrict__ emb,
                                               float* __restrict__ x) {
  const int row = blockIdx.x;
  const float* e = emb + (size_t)inp[row] * Dd;
  float* o = x + (size_t)row * Dd;
  for (int i = threadIdx.x; i < Dd; i += 256) o[i] = e[i];
}

// ---------------- WMMA GEMM: C = act(A @ W^T + bias), bf16 in, f32 out ------
// Block = 4 waves; wave computes 16 rows x 64 cols (4 accumulators, A reuse x4).
// Software-pipelined: fragments for k-chunk n+1 are in flight while chunk n's
// four WMMAs execute, so waits are partial instead of full drains.
__global__ __launch_bounds__(128) void k_gemm_nt(const unsigned short* __restrict__ A,
                                                 const unsigned short* __restrict__ W,
                                                 const float* __restrict__ bias,
                                                 float* __restrict__ C,
                                                 int M, int N, int Kd, int relu) {
  const int wave = threadIdx.x >> 5;
  const int lane = threadIdx.x & 31;
  const int half = lane >> 4;
  const int l16  = lane & 15;
  const int rowbase = blockIdx.y * 16;
  const int colbase = blockIdx.x * 256 + wave * 64;

  const v8f z = {0.f, 0.f, 0.f, 0.f, 0.f, 0.f, 0.f, 0.f};
  v8f acc[4] = {z, z, z, z};

  int arow = rowbase + l16; if (arow > M - 1) arow = M - 1;
  const unsigned short* Ap = A + (size_t)arow * Kd + half * 8;
  const unsigned short* Wr[4];
#pragma unroll
  for (int j = 0; j < 4; ++j) {
    int n = colbase + j * 16 + l16; if (n > N - 1) n = N - 1;
    Wr[j] = W + (size_t)n * Kd + half * 8;
  }

  auto loadA = [&](int kk, Frag& a) {
    a.q[0] = *reinterpret_cast<const uint4*>(Ap + kk);        // K 0..7   (half)
    a.q[1] = *reinterpret_cast<const uint4*>(Ap + kk + 16);   // K 16..23 (half)
  };
  auto loadB = [&](int kk, Frag* b) {
#pragma unroll
    for (int j = 0; j < 4; ++j) {
      b[j].q[0] = *reinterpret_cast<const uint4*>(Wr[j] + kk);
      b[j].q[1] = *reinterpret_cast<const uint4*>(Wr[j] + kk + 16);
    }
  };
  auto mm4 = [&](const Frag& a, const Frag* b) {
#pragma unroll
    for (int j = 0; j < 4; ++j)
      acc[j] = __builtin_amdgcn_wmma_f32_16x16x32_bf16(
          false, a.v, false, b[j].v, (short)0, acc[j], false, false);
  };

  // Kd is always a multiple of 64 -> even number of 32-deep chunks.
  Frag aA, aB; Frag bA[4], bB[4];
  loadA(0, aA); loadB(0, bA);
  int kk = 0;
  for (; kk < Kd - 64; kk += 64) {
    loadA(kk + 32, aB); loadB(kk + 32, bB);
    mm4(aA, bA);
    loadA(kk + 64, aA); loadB(kk + 64, bA);
    mm4(aB, bB);
  }
  loadA(kk + 32, aB); loadB(kk + 32, bB);
  mm4(aA, bA);
  mm4(aB, bB);

#pragma unroll
  for (int j = 0; j < 4; ++j) {
    const int gcol = colbase + j * 16 + l16;
    if (gcol >= N) continue;
    const float bv = bias ? bias[gcol] : 0.f;
#pragma unroll
    for (int r = 0; r < 8; ++r) {
      const int grow = rowbase + half * 8 + r;
      if (grow >= M) continue;
      float v = acc[j][r] + bv;
      if (relu) v = fmaxf(v, 0.f);
      C[(size_t)grow * N + gcol] = v;
    }
  }
}

// ---------------- attention: one wave per (b,h,q-row) -----------------------
__global__ __launch_bounds__(128) void k_attn(const float* __restrict__ qkv,
                                              float* __restrict__ out,
                                              int lastOnly) {
  __shared__ float qsh[4][64];
  __shared__ float ash[4][256];
  const int wave = threadIdx.x >> 5;
  const int lane = threadIdx.x & 31;
  const int w = blockIdx.x * 4 + wave;
  int b, h, s;
  if (lastOnly) { h = w & (Hh - 1); b = w >> 4; s = Ss - 1; }
  else          { s = w & (Ss - 1); h = (w >> 8) & (Hh - 1); b = w >> 12; }
  const int qrow = b * Ss + s;
  const float* qp = qkv + (size_t)qrow * (3 * Dd) + h * 64;
  qsh[wave][lane]      = qp[lane];
  qsh[wave][lane + 32] = qp[lane + 32];
  __syncthreads();

  const float* kbase = qkv + (size_t)(b * Ss) * (3 * Dd) + Dd + h * 64;
  float sc[8];
#pragma unroll
  for (int jj = 0; jj < 8; ++jj) {
    const float* kp = kbase + (size_t)(lane + 32 * jj) * (3 * Dd);
    float a = 0.f;
    for (int d = 0; d < 64; ++d) a += qsh[wave][d] * kp[d];
    sc[jj] = a * 0.125f;              // 1/sqrt(64)
  }
  float m = sc[0];
#pragma unroll
  for (int jj = 1; jj < 8; ++jj) m = fmaxf(m, sc[jj]);
  for (int off = 16; off > 0; off >>= 1) m = fmaxf(m, __shfl_xor(m, off, 32));
  float sum = 0.f;
#pragma unroll
  for (int jj = 0; jj < 8; ++jj) { sc[jj] = __expf(sc[jj] - m); sum += sc[jj]; }
  for (int off = 16; off > 0; off >>= 1) sum += __shfl_xor(sum, off, 32);
  const float inv = 1.f / sum;
#pragma unroll
  for (int jj = 0; jj < 8; ++jj) ash[wave][lane + 32 * jj] = sc[jj] * inv;
  __syncthreads();

  const float* vbase = qkv + (size_t)(b * Ss) * (3 * Dd) + 2 * Dd + h * 64;
  const int d0 = lane * 2;
  float o0 = 0.f, o1 = 0.f;
  for (int k2 = 0; k2 < Ss; ++k2) {
    const float a = ash[wave][k2];
    const float* vp = vbase + (size_t)k2 * (3 * Dd);
    o0 += a * vp[d0];
    o1 += a * vp[d0 + 1];
  }
  const int orow = lastOnly ? b : qrow;
  out[(size_t)orow * Dd + h * 64 + d0]     = o0;
  out[(size_t)orow * Dd + h * 64 + d0 + 1] = o1;
}

// ---------------- residual + LayerNorm (row per block) ----------------------
static __device__ __forceinline__ float block_sum(float v, float* red) {
  const int tid = threadIdx.x;
  red[tid] = v;
  __syncthreads();
  for (int s = 128; s > 0; s >>= 1) {
    if (tid < s) red[tid] += red[tid + s];
    __syncthreads();
  }
  const float r = red[0];
  __syncthreads();
  return r;
}

__global__ __launch_bounds__(256) void k_add_ln(const float* __restrict__ a,
                                                const float* __restrict__ b,
                                                const float* __restrict__ g,
                                                const float* __restrict__ beta,
                                                float* __restrict__ y, float scale) {
  __shared__ float red[256];
  const int row = blockIdx.x, tid = threadIdx.x;
  const float* pa = a + (size_t)row * Dd;
  const float* pb = b + (size_t)row * Dd;
  float v[4]; float s = 0.f;
#pragma unroll
  for (int i = 0; i < 4; ++i) { v[i] = pa[tid + i * 256] + pb[tid + i * 256]; s += v[i]; }
  const float mean = block_sum(s, red) * (1.f / Dd);
  float q = 0.f;
#pragma unroll
  for (int i = 0; i < 4; ++i) { const float d = v[i] - mean; q += d * d; }
  const float var = block_sum(q, red) * (1.f / Dd);
  const float inv = rsqrtf(var + 1e-5f);
  float* py = y + (size_t)row * Dd;
#pragma unroll
  for (int i = 0; i < 4; ++i) {
    const int c = tid + i * 256;
    py[c] = ((v[i] - mean) * inv * g[c] + beta[c]) * scale;
  }
}

// ---------------- gather last-position rows ---------------------------------
__global__ __launch_bounds__(256) void k_last_rows(const float* __restrict__ in,
                                                   float* __restrict__ out) {
  const int i = blockIdx.x * 256 + threadIdx.x;   // Bb*Dd
  const int b = i >> 10, d = i & (Dd - 1);
  out[i] = in[((size_t)b * Ss + (Ss - 1)) * Dd + d];
}

// ---------------- gate: weights[b,n] = ctx_last[b]·gate_w[n] + gate_b[n] ----
__global__ __launch_bounds__(256) void k_gate(const float* __restrict__ ctxl,
                                              const float* __restrict__ gw,
                                              const float* __restrict__ gb,
                                              float* __restrict__ wout) {
  const int n = blockIdx.x * 256 + threadIdx.x;
  if (n >= Nn) return;
  const float* wr = gw + (size_t)n * Dd;
  float acc[Bb] = {0.f, 0.f, 0.f, 0.f};
  for (int d = 0; d < Dd; ++d) {
    const float wv = wr[d];
#pragma unroll
    for (int b = 0; b < Bb; ++b) acc[b] += ctxl[b * Dd + d] * wv;
  }
  const float bv = gb[n];
#pragma unroll
  for (int b = 0; b < Bb; ++b) wout[b * Nn + n] = acc[b] + bv;
}

// ---------------- gate reduce: fm, std, total, importance, top-k, softmax ---
__global__ __launch_bounds__(256) void k_gate_reduce(const float* __restrict__ wts,
                                                     const float* __restrict__ noise,
                                                     float* __restrict__ jw,
                                                     float* __restrict__ imp) {
  __shared__ float tot[Nn];
  __shared__ float red[256];
  const int tid = threadIdx.x;
  for (int n = tid; n < Nn; n += 256) {
    float s = 0.f;
    for (int b = 0; b < Bb; ++b) s += wts[b * Nn + n];
    tot[n] = s * (1.f / Bb);
  }
  __syncthreads();
  float ls = 0.f;
  for (int n = tid; n < Nn; n += 256) ls += tot[n];
  const float mean_fm = block_sum(ls, red) * (1.f / Nn);
  float lq = 0.f;
  for (int n = tid; n < Nn; n += 256) { const float d = tot[n] - mean_fm; lq += d * d; }
  const float std_fm = sqrtf(block_sum(lq, red) * (1.f / (Nn - 1)));
  for (int n = tid; n < Nn; n += 256) tot[n] += noise[n] * std_fm;
  __syncthreads();
  ls = 0.f;
  for (int n = tid; n < Nn; n += 256) ls += tot[n];
  const float mean_t = block_sum(ls, red) * (1.f / Nn);
  lq = 0.f;
  for (int n = tid; n < Nn; n += 256) { const float d = tot[n] - mean_t; lq += d * d; }
  const float std_t = sqrtf(block_sum(lq, red) * (1.f / (Nn - 1)));
  if (tid == 0) {
    const float r = std_t / mean_t;
    imp[0] = 0.1f * r * r;
    float top[Kk];
#pragma unroll
    for (int i = 0; i < Kk; ++i) top[i] = NEG_INF;
    for (int n = 0; n < Nn; ++n) {          // insertion-select top-K descending
      const float v = tot[n];
      if (v > top[Kk - 1]) {
        int i = Kk - 1;
        while (i > 0 && top[i - 1] < v) { top[i] = top[i - 1]; --i; }
        top[i] = v;
      }
    }
    const float mx = top[0];
    float se = 0.f;
    for (int i = 0; i < Kk; ++i) { top[i] = __expf(top[i] - mx); se += top[i]; }
    const float is = 1.f / se;
    for (int i = 0; i < Kk; ++i) jw[i] = top[i] * is;
  }
}

// ---------------- mix responses: x = sum_k jw[k]*responses[k] ---------------
__global__ __launch_bounds__(256) void k_mix(const float* __restrict__ resp,
                                             const float* __restrict__ jw,
                                             float* __restrict__ x) {
  __shared__ float w[Kk];
  if (threadIdx.x < Kk) w[threadIdx.x] = jw[threadIdx.x];
  __syncthreads();
  const int i = blockIdx.x * 256 + threadIdx.x;
  float a = 0.f;
#pragma unroll
  for (int k = 0; k < Kk; ++k) a += w[k] * resp[(size_t)k * (BSr * Dd) + i];
  x[i] = a;
}

// ---------------- fused decoder GEMM + online log-softmax + CE --------------
// Block: 16 activation rows (bf16 A tile in LDS). 8 waves sweep the
// zero-padded bf16 vocab matrix in 128-column strips; software-pipelined
// ds/b128 + wmma inner loop; per-row online (max, sumexp) via half-wave
// shuffles, cross-wave combine in LDS.
__global__ __launch_bounds__(256) void k_dec_ce(const unsigned short* __restrict__ Xb,
                                                const unsigned short* __restrict__ Wdb,
                                                const int* __restrict__ inp,
                                                float* __restrict__ ce) {
  __shared__ unsigned short Abf[16 * Dd];   // 32 KB bf16 A tile
  __shared__ float wm[8][16], wsm[8][16], wlb[8][16];
  const int tid  = threadIdx.x;
  const int wave = tid >> 5, lane = tid & 31, half = lane >> 4, l16 = lane & 15;
  const int rowbase = blockIdx.x * 16;

  // straight bf16 copy: rows are contiguous in Xb
  {
    const uint4* src = reinterpret_cast<const uint4*>(Xb + (size_t)rowbase * Dd);
    uint4* dst = reinterpret_cast<uint4*>(Abf);
    for (int i = tid; i < 16 * Dd / 8; i += 256) dst[i] = src[i];
  }
  __syncthreads();

  int lblr[8];
#pragma unroll
  for (int r = 0; r < 8; ++r) {
    const int grow = rowbase + half * 8 + r;
    const int b = grow >> 8, s = grow & (Ss - 1);
    lblr[r] = (s < Ss - 1) ? inp[b * Ss + s + 1] : -1;
  }

  float rm[8], rs[8], lab[8];
#pragma unroll
  for (int r = 0; r < 8; ++r) { rm[r] = NEG_INF; rs[r] = 0.f; lab[r] = NEG_INF; }

  const unsigned short* Ap = Abf + l16 * Dd + half * 8;

  const int nchunk = VvP >> 7;   // 393 strips of 128 vocab columns
  for (int c = 0; c < nchunk; ++c) {
    const int colbase = c * 128 + wave * 16;
    const int coln = colbase + l16;
    if (colbase + 128 + l16 < VvP)
      __builtin_prefetch(Wdb + (size_t)(colbase + 128 + l16) * Dd, 0, 1);

    const unsigned short* Wp = Wdb + (size_t)coln * Dd + half * 8;

    auto loadF = [&](int kk, Frag& a, Frag& b) {
      a.q[0] = *reinterpret_cast<const uint4*>(Ap + kk);
      a.q[1] = *reinterpret_cast<const uint4*>(Ap + kk + 16);
      b.q[0] = *reinterpret_cast<const uint4*>(Wp + kk);
      b.q[1] = *reinterpret_cast<const uint4*>(Wp + kk + 16);
    };

    v8f acc = {0.f, 0.f, 0.f, 0.f, 0.f, 0.f, 0.f, 0.f};
    Frag aA, bA, aB, bB;
    loadF(0, aA, bA);
    int kk = 0;
    for (; kk < Dd - 64; kk += 64) {
      loadF(kk + 32, aB, bB);
      acc = __builtin_amdgcn_wmma_f32_16x16x32_bf16(
          false, aA.v, false, bA.v, (short)0, acc, false, false);
      loadF(kk + 64, aA, bA);
      acc = __builtin_amdgcn_wmma_f32_16x16x32_bf16(
          false, aB.v, false, bB.v, (short)0, acc, false, false);
    }
    loadF(kk + 32, aB, bB);
    acc = __builtin_amdgcn_wmma_f32_16x16x32_bf16(
        false, aA.v, false, bA.v, (short)0, acc, false, false);
    acc = __builtin_amdgcn_wmma_f32_16x16x32_bf16(
        false, aB.v, false, bB.v, (short)0, acc, false, false);

#pragma unroll
    for (int r = 0; r < 8; ++r) {
      float t = (coln < Vv) ? acc[r] : NEG_INF;
      if (coln == lblr[r]) lab[r] = t;
      float mx = t;
      for (int mk = 1; mk <= 8; mk <<= 1) mx = fmaxf(mx, __shfl_xor(mx, mk, 32));
      const float nm = fmaxf(rm[r], mx);
      float p = __expf(t - nm);
      for (int mk = 1; mk <= 8; mk <<= 1) p += __shfl_xor(p, mk, 32);
      rs[r] = rs[r] * __expf(rm[r] - nm) + p;
      rm[r] = nm;
    }
  }
#pragma unroll
  for (int r = 0; r < 8; ++r) {
    float lv = lab[r];
    for (int mk = 1; mk <= 8; mk <<= 1) lv = fmaxf(lv, __shfl_xor(lv, mk, 32));
    if (l16 == 0) {
      wm[wave][half * 8 + r]  = rm[r];
      wsm[wave][half * 8 + r] = rs[r];
      wlb[wave][half * 8 + r] = lv;
    }
  }
  __syncthreads();
  if (tid < 16) {
    const int grow = rowbase + tid;
    const int s = grow & (Ss - 1);
    if (s < Ss - 1) {
      float Mv = NEG_INF;
      for (int w2 = 0; w2 < 8; ++w2) Mv = fmaxf(Mv, wm[w2][tid]);
      float Sv = 0.f, Lv = NEG_INF;
      for (int w2 = 0; w2 < 8; ++w2) {
        Sv += wsm[w2][tid] * __expf(wm[w2][tid] - Mv);
        Lv = fmaxf(Lv, wlb[w2][tid]);
      }
      const float logp = Lv - Mv - __logf(Sv);
      atomicAdd(ce, -logp);
    }
  }
}

__global__ void k_zero(float* p) { if (threadIdx.x == 0) p[0] = 0.f; }

__global__ void k_final(const float* ce, const float* imp, float* out) {
  if (threadIdx.x == 0) out[0] = ce[0] * (1.f / (float)(Bb * (Ss - 1))) + imp[0];
}

// ---------------------------------------------------------------------------
extern "C" void kernel_launch(void* const* d_in, const int* in_sizes, int n_in,
                              void* d_out, int out_size, void* d_ws, size_t ws_size,
                              hipStream_t stream) {
  (void)in_sizes; (void)n_in; (void)out_size; (void)ws_size;
  const int*   inp   = (const int*)  d_in[0];
  const float* resp  = (const float*)d_in[1];
  const float* noise = (const float*)d_in[2];
  const float* emb   = (const float*)d_in[3];
  const float* lWqkv = (const float*)d_in[4];
  const float* lbqkv = (const float*)d_in[5];
  const float* lWo   = (const float*)d_in[6];
  const float* lbo   = (const float*)d_in[7];
  const float* lg1   = (const float*)d_in[8];
  const float* lb1   = (const float*)d_in[9];
  const float* lW1   = (const float*)d_in[10];
  const float* lbb1  = (const float*)d_in[11];
  const float* lW2   = (const float*)d_in[12];
  const float* lbb2  = (const float*)d_in[13];
  const float* lg2   = (const float*)d_in[14];
  const float* lb2   = (const float*)d_in[15];
  const float* eWqkv = (const float*)d_in[16];
  const float* ebqkv = (const float*)d_in[17];
  const float* eWo   = (const float*)d_in[18];
  const float* ebo   = (const float*)d_in[19];
  const float* eg1   = (const float*)d_in[20];
  const float* eb1   = (const float*)d_in[21];
  const float* eW1   = (const float*)d_in[22];
  const float* ebb1  = (const float*)d_in[23];
  const float* eW2   = (const float*)d_in[24];
  const float* ebb2  = (const float*)d_in[25];
  const float* eg2   = (const float*)d_in[26];
  const float* eb2   = (const float*)d_in[27];
  const float* gw    = (const float*)d_in[28];
  const float* gb    = (const float*)d_in[29];
  const float* decw  = (const float*)d_in[30];

  float* out  = (float*)d_out;
  float* wout = out + 1;                 // weights (B,N) after the scalar

  // ---- f32 workspace ----
  float* ws = (float*)d_ws;
  float* x    = ws; ws += (size_t)BSr * Dd;
  float* x1   = ws; ws += (size_t)BSr * Dd;
  float* qkv  = ws; ws += (size_t)BSr * 3 * Dd;
  float* ob   = ws; ws += (size_t)BSr * Dd;
  float* tmp  = ws; ws += (size_t)BSr * Dd;
  float* hbuf = ws; ws += (size_t)BSr * FFd;
  float* xl   = ws; ws += Bb * Dd;
  float* ol   = ws; ws += Bb * Dd;
  float* tmpl = ws; ws += Bb * Dd;
  float* x1l  = ws; ws += Bb * Dd;
  float* hl   = ws; ws += Bb * FFd;
  float* fl   = ws; ws += Bb * Dd;
  float* ctxl = ws; ws += Bb * Dd;
  float* jw   = ws; ws += 32;
  float* imp  = ws; ws += 1;
  float* ce   = ws; ws += 1;
  ws += 30;                               // pad to 16B boundary

  // ---- bf16 workspace ----
  unsigned short* us = (unsigned short*)ws;
  unsigned short* abf    = us; us += (size_t)BSr * FFd;       // activation scratch (max M*K)
  unsigned short* lWqkvB = us; us += (size_t)3 * Dd * Dd;
  unsigned short* lWoB   = us; us += (size_t)Dd * Dd;
  unsigned short* lW1B   = us; us += (size_t)FFd * Dd;
  unsigned short* lW2B   = us; us += (size_t)Dd * FFd;
  unsigned short* eWqkvB = us; us += (size_t)2 * 3 * Dd * Dd;
  unsigned short* eWoB   = us; us += (size_t)2 * Dd * Dd;
  unsigned short* eW1B   = us; us += (size_t)2 * FFd * Dd;
  unsigned short* eW2B   = us; us += (size_t)2 * Dd * FFd;
  unsigned short* decB   = us; us += (size_t)VvP * Dd;

  const dim3 blk128(128), blk256(256);
  auto cvt = [&](const float* s, unsigned short* dst, size_t n, size_t ntot) {
    const int n4t = (int)(ntot / 4);
    k_cvt<<<(n4t + 255) / 256, blk256, 0, stream>>>(s, dst, (int)(n / 4), n4t);
  };

  // ---- one-time (per call) weight conversions to bf16 ----
  cvt(lWqkv, lWqkvB, (size_t)3 * Dd * Dd, (size_t)3 * Dd * Dd);
  cvt(lWo,   lWoB,   (size_t)Dd * Dd,     (size_t)Dd * Dd);
  cvt(lW1,   lW1B,   (size_t)FFd * Dd,    (size_t)FFd * Dd);
  cvt(lW2,   lW2B,   (size_t)Dd * FFd,    (size_t)Dd * FFd);
  cvt(eWqkv, eWqkvB, (size_t)2 * 3 * Dd * Dd, (size_t)2 * 3 * Dd * Dd);
  cvt(eWo,   eWoB,   (size_t)2 * Dd * Dd, (size_t)2 * Dd * Dd);
  cvt(eW1,   eW1B,   (size_t)2 * FFd * Dd, (size_t)2 * FFd * Dd);
  cvt(eW2,   eW2B,   (size_t)2 * Dd * FFd, (size_t)2 * Dd * FFd);
  cvt(decw,  decB,   (size_t)Vv * Dd,     (size_t)VvP * Dd);   // zero-padded rows

  // ---- loc encoder layer (full QKV, then only last-position rows) ----
  k_embed<<<BSr, blk256, 0, stream>>>(inp, emb, x);
  cvt(x, abf, (size_t)BSr * Dd, (size_t)BSr * Dd);
  k_gemm_nt<<<dim3(3 * Dd / 256, BSr / 16), blk128, 0, stream>>>(abf, lWqkvB, lbqkv, qkv, BSr, 3 * Dd, Dd, 0);
  k_attn<<<Bb * Hh / 4, blk128, 0, stream>>>(qkv, ol, 1);
  cvt(ol, abf, Bb * Dd, Bb * Dd);
  k_gemm_nt<<<dim3(Dd / 256, 1), blk128, 0, stream>>>(abf, lWoB, lbo, tmpl, Bb, Dd, Dd, 0);
  k_last_rows<<<Bb * Dd / 256, blk256, 0, stream>>>(x, xl);
  k_add_ln<<<Bb, blk256, 0, stream>>>(xl, tmpl, lg1, lb1, x1l, 1.f);
  cvt(x1l, abf, Bb * Dd, Bb * Dd);
  k_gemm_nt<<<dim3(FFd / 256, 1), blk128, 0, stream>>>(abf, lW1B, lbb1, hl, Bb, FFd, Dd, 1);
  cvt(hl, abf, Bb * FFd, Bb * FFd);
  k_gemm_nt<<<dim3(Dd / 256, 1), blk128, 0, stream>>>(abf, lW2B, lbb2, fl, Bb, Dd, FFd, 0);
  k_add_ln<<<Bb, blk256, 0, stream>>>(x1l, fl, lg2, lb2, ctxl, 32.f);   // *sqrt(D)

  // ---- gating, top-k softmax, response mixing ----
  k_gate<<<(Nn + 255) / 256, blk256, 0, stream>>>(ctxl, gw, gb, wout);
  k_gate_reduce<<<1, blk256, 0, stream>>>(wout, noise, jw, imp);
  k_mix<<<BSr * Dd / 256, blk256, 0, stream>>>(resp, jw, x);

  // ---- 2 encoder layers ----
  for (int l = 0; l < 2; ++l) {
    cvt(x, abf, (size_t)BSr * Dd, (size_t)BSr * Dd);
    k_gemm_nt<<<dim3(3 * Dd / 256, BSr / 16), blk128, 0, stream>>>(
        abf, eWqkvB + (size_t)l * 3 * Dd * Dd, ebqkv + l * 3 * Dd, qkv, BSr, 3 * Dd, Dd, 0);
    k_attn<<<Bb * Hh * Ss / 4, blk128, 0, stream>>>(qkv, ob, 0);
    cvt(ob, abf, (size_t)BSr * Dd, (size_t)BSr * Dd);
    k_gemm_nt<<<dim3(Dd / 256, BSr / 16), blk128, 0, stream>>>(
        abf, eWoB + (size_t)l * Dd * Dd, ebo + l * Dd, tmp, BSr, Dd, Dd, 0);
    k_add_ln<<<BSr, blk256, 0, stream>>>(x, tmp, eg1 + l * Dd, eb1 + l * Dd, x1, 1.f);
    cvt(x1, abf, (size_t)BSr * Dd, (size_t)BSr * Dd);
    k_gemm_nt<<<dim3(FFd / 256, BSr / 16), blk128, 0, stream>>>(
        abf, eW1B + (size_t)l * FFd * Dd, ebb1 + l * FFd, hbuf, BSr, FFd, Dd, 1);
    cvt(hbuf, abf, (size_t)BSr * FFd, (size_t)BSr * FFd);
    k_gemm_nt<<<dim3(Dd / 256, BSr / 16), blk128, 0, stream>>>(
        abf, eW2B + (size_t)l * Dd * FFd, ebb2 + l * Dd, tmp, BSr, Dd, FFd, 0);
    k_add_ln<<<BSr, blk256, 0, stream>>>(x1, tmp, eg2 + l * Dd, eb2 + l * Dd, x, 1.f);
  }

  // ---- fused decoder + cross-entropy ----
  cvt(x, abf, (size_t)BSr * Dd, (size_t)BSr * Dd);
  k_zero<<<1, 1, 0, stream>>>(ce);
  k_dec_ce<<<BSr / 16, blk256, 0, stream>>>(abf, decB, inp, ce);
  k_final<<<1, 1, 0, stream>>>(ce, imp, out);
}
